// AnomalyAttention_51994874085548
// MI455X (gfx1250) — compile-verified, matrix operand order checked by reading, not run on hardware
//
#include <hip/hip_runtime.h>
#include <hip/hip_bf16.h>
#include <math.h>

typedef __bf16 bf16_t;
typedef __attribute__((ext_vector_type(16))) __bf16 v16bf;
typedef __attribute__((ext_vector_type(8)))  __bf16 bf16x8;
typedef __attribute__((ext_vector_type(4)))  __bf16 bf16x4;
typedef __attribute__((ext_vector_type(8)))  float   f32x8;
typedef __attribute__((ext_vector_type(8)))  float   v8f;
typedef __attribute__((ext_vector_type(4)))  float   f32x4;

static constexpr int kL = 512;     // sequence length (win_size+1)
static constexpr int kE = 128;     // head dim
static constexpr int kHeads = 128; // B*C*H = 4*4*8
static constexpr int kTM = 32;     // rows per block tile
static constexpr int kNC = 128;    // key/value rows staged per LDS chunk

// Assemble a 16-bit WMMA fragment from LDS (row-major, 'rowbase' points at the
// lane's matrix row). Element e of the v16bf maps to K = k0 + e + (e&8) + 8*hi
// per the CDNA5 16-bit A/B VGPR layout -> two contiguous 16-byte LDS loads.
__device__ __forceinline__ v16bf load_frag_bf16(const bf16_t* rowbase, int k0, int hi) {
  bf16x8 lo = *(const bf16x8*)(rowbase + k0 + 8 * hi);
  bf16x8 hh = *(const bf16x8*)(rowbase + k0 + 16 + 8 * hi);
  return __builtin_shufflevector(lo, hh, 0,1,2,3,4,5,6,7,8,9,10,11,12,13,14,15);
}

// Same fragment pattern but source data is f32 in LDS; convert to bf16 in regs.
__device__ __forceinline__ v16bf load_frag_f32(const float* rowbase, int k0, int hi) {
  f32x8 lo = *(const f32x8*)(rowbase + k0 + 8 * hi);
  f32x8 hh = *(const f32x8*)(rowbase + k0 + 16 + 8 * hi);
  v16bf a;
#pragma unroll
  for (int j = 0; j < 8; ++j) { a[j] = (__bf16)lo[j]; a[j + 8] = (__bf16)hh[j]; }
  return a;
}

__global__ __launch_bounds__(256) void anomaly_attn_kernel(
    const float* __restrict__ Q, const float* __restrict__ K,
    const float* __restrict__ V, const float* __restrict__ Sg,
    float* __restrict__ outV, float* __restrict__ outSeries,
    float* __restrict__ outPrior, float* __restrict__ outSigma) {
  __shared__ float  Slds[kTM * kL];      // 64 KB: score tile (f32)
  __shared__ bf16_t Qs[kTM * kE];        //  8 KB: Q tile (bf16)
  __shared__ bf16_t KVs[kNC * kE];       // 32 KB: K chunk (row-major) / V chunk (transposed)
  __shared__ float  sigs[kTM];
  __shared__ float  red[256];
  __shared__ float  rowstat[kTM];

  const int tid  = threadIdx.x;
  const int head = blockIdx.x >> 4;            // 128 heads
  const int m0   = (blockIdx.x & 15) * kTM;    // 16 row tiles per head
  const int wave = tid >> 5;
  const int lane = tid & 31;
  const int l15  = lane & 15;
  const int hi   = lane >> 4;
  const int strip = wave & 1;                  // which 16-row strip of the 32-row tile
  const int quad  = wave >> 1;                 // which 32-col slice (of 128) per chunk

  const float* Qh = Q + (size_t)head * kL * kE;
  const float* Kh = K + (size_t)head * kL * kE;
  const float* Vh = V + (size_t)head * kL * kE;

  // ---- Phase 0: stage Q tile to bf16 LDS; transform sigma per row ----------
  {
    const f32x4* Qg = (const f32x4*)(Qh + (size_t)m0 * kE);
    for (int i = tid; i < kTM * kE / 4; i += 256) {
      f32x4 q = Qg[i];
      bf16x4 b;
      b[0] = (__bf16)q[0]; b[1] = (__bf16)q[1];
      b[2] = (__bf16)q[2]; b[3] = (__bf16)q[3];
      ((bf16x4*)Qs)[i] = b;
    }
    if (tid < kTM) {
      float x = Sg[(size_t)head * kL + m0 + tid];
      float s = 1.0f / (1.0f + __expf(-5.0f * x)) + 1e-5f;  // sigmoid(5x)+1e-5
      s = exp2f(s * 1.5849625007211562f) - 1.0f;            // 3^s - 1
      sigs[tid] = s;
    }
  }
  __syncthreads();

  // ---- Gaussian prior + sigma_out (independent elementwise, store early) ---
  {
    const float c0 = 0.3989422804014327f;  // 1/sqrt(2*pi)
    const size_t base = ((size_t)head * kL + m0) * kL;
    for (int i = tid; i < kTM * kL; i += 256) {
      int m = i >> 9, n = i & (kL - 1);
      float s = sigs[m];
      float d = fabsf((float)((m0 + m) - n));
      float p = (c0 / s) * __expf(-(d * d) / (2.0f * s * s));
      outPrior[base + i] = p;
      outSigma[base + i] = s;
    }
  }

  // ---- Phase 1: S = scale * Q K^T with causal mask -------------------------
  const float scale = 0.08838834764831845f;  // 1/sqrt(128)
  for (int nc = 0; nc < kL / kNC; ++nc) {
    // stage K rows [nc*kNC, +kNC) as bf16 row-major
    const f32x4* Kg = (const f32x4*)(Kh + (size_t)nc * kNC * kE);
    for (int i = tid; i < kNC * kE / 4; i += 256) {
      f32x4 k4 = Kg[i];
      bf16x4 b;
      b[0] = (__bf16)k4[0]; b[1] = (__bf16)k4[1];
      b[2] = (__bf16)k4[2]; b[3] = (__bf16)k4[3];
      ((bf16x4*)KVs)[i] = b;
    }
    __syncthreads();
#pragma unroll
    for (int nt = 0; nt < 2; ++nt) {
      const int nloc = quad * 32 + nt * 16;  // col base within chunk
      v8f acc = {};
      const bf16_t* arow = Qs + (strip * 16 + l15) * kE;       // A: query row
      const bf16_t* brow = KVs + (nloc + l15) * kE;            // B: key row (N=lane&15)
#pragma unroll
      for (int k0 = 0; k0 < kE; k0 += 32) {
        v16bf a = load_frag_bf16(arow, k0, hi);
        v16bf b = load_frag_bf16(brow, k0, hi);
        acc = __builtin_amdgcn_wmma_f32_16x16x32_bf16(false, a, false, b,
                                                      (short)0, acc, false, false);
      }
      const int nabs = nc * kNC + nloc + l15;
#pragma unroll
      for (int r = 0; r < 8; ++r) {
        const int mloc = strip * 16 + r + 8 * hi;  // C/D layout: M = r + 8*(lane>=16)
        float v = acc[r] * scale;
        if (nabs > m0 + mloc) v = -__builtin_inff();  // causal mask
        Slds[mloc * kL + nabs] = v;
      }
    }
    __syncthreads();
  }

  // ---- Phase 2: row softmax (fp32), write series, keep probs in LDS --------
  {
    const int row = tid >> 3, part = tid & 7;
    float* srow = Slds + row * kL + part * 64;
    float lmax = -__builtin_inff();
    for (int j = 0; j < 64; ++j) lmax = fmaxf(lmax, srow[j]);
    red[tid] = lmax;
    __syncthreads();
    if (part == 0) {
      float m = red[tid];
      for (int j = 1; j < 8; ++j) m = fmaxf(m, red[tid + j]);
      rowstat[row] = m;
    }
    __syncthreads();
    const float rm = rowstat[row];
    float lsum = 0.0f;
    for (int j = 0; j < 64; ++j) {
      float e = __expf(srow[j] - rm);  // masked entries -> exp(-inf) = 0
      srow[j] = e;
      lsum += e;
    }
    red[tid] = lsum;
    __syncthreads();
    if (part == 0) {
      float s = 0.0f;
      for (int j = 0; j < 8; ++j) s += red[tid + j];
      rowstat[row] = 1.0f / s;
    }
    __syncthreads();
    const float rinv = rowstat[row];
    float* gout = outSeries + ((size_t)head * kL + m0 + row) * kL + part * 64;
    for (int j = 0; j < 64; j += 4) {
      f32x4 v = *(f32x4*)(srow + j);
      v *= rinv;
      *(f32x4*)(srow + j) = v;
      *(f32x4*)(gout + j) = v;
    }
  }
  __syncthreads();

  // ---- Phase 3: Vout = series @ V ------------------------------------------
  v8f vacc[2] = {{}, {}};
  for (int kc = 0; kc < kL / kNC; ++kc) {
    // stage V rows [kc*kNC, +kNC) TRANSPOSED: KVs[ecol*kNC + keyrow] (bf16)
    {
      const f32x4* Vg = (const f32x4*)(Vh + (size_t)kc * kNC * kE);
      for (int i = tid; i < kNC * kE / 4; i += 256) {
        f32x4 v4 = Vg[i];
        const int krow = i >> 5;           // 32 float4 per 128-wide row
        const int e0 = (i & 31) * 4;
#pragma unroll
        for (int j = 0; j < 4; ++j) KVs[(e0 + j) * kNC + krow] = (__bf16)v4[j];
      }
    }
    __syncthreads();
#pragma unroll
    for (int nt = 0; nt < 2; ++nt) {
      const float* arow = Slds + (strip * 16 + l15) * kL + kc * kNC;  // probs (f32)
      const int ecol = quad * 32 + nt * 16 + l15;
      const bf16_t* brow = KVs + ecol * kNC;                          // V^T row
#pragma unroll
      for (int k0 = 0; k0 < kNC; k0 += 32) {
        v16bf a = load_frag_f32(arow, k0, hi);
        v16bf b = load_frag_bf16(brow, k0, hi);
        vacc[nt] = __builtin_amdgcn_wmma_f32_16x16x32_bf16(false, a, false, b,
                                                           (short)0, vacc[nt], false, false);
      }
    }
    __syncthreads();
  }
#pragma unroll
  for (int nt = 0; nt < 2; ++nt) {
    const int ecol = quad * 32 + nt * 16 + l15;
#pragma unroll
    for (int r = 0; r < 8; ++r) {
      const int mloc = strip * 16 + r + 8 * hi;
      outV[((size_t)head * kL + m0 + mloc) * kE + ecol] = vacc[nt][r];
    }
  }
}

extern "C" void kernel_launch(void* const* d_in, const int* in_sizes, int n_in,
                              void* d_out, int out_size, void* d_ws, size_t ws_size,
                              hipStream_t stream) {
  const float* Q  = (const float*)d_in[0];
  const float* K  = (const float*)d_in[1];
  const float* V  = (const float*)d_in[2];
  const float* Sg = (const float*)d_in[3];
  // d_in[4] is the boolean causal mask; it is regenerated analytically in-kernel.

  float* out = (float*)d_out;
  float* outV      = out;                                    // [128,512,128]
  float* outSeries = outV + (size_t)kHeads * kL * kE;        // [128,512,512]
  float* outPrior  = outSeries + (size_t)kHeads * kL * kL;   // [128,512,512]
  float* outSigma  = outPrior + (size_t)kHeads * kL * kL;    // [128,512,512]

  dim3 grid(kHeads * (kL / kTM));  // 128 heads * 16 row tiles = 2048 blocks
  dim3 block(256);                 // 8 wave32 waves
  hipLaunchKernelGGL(anomaly_attn_kernel, grid, block, 0, stream,
                     Q, K, V, Sg, outV, outSeries, outPrior, outSigma);
}
